// SingleCPGNN_64132451664068
// MI455X (gfx1250) — matrix operand order repaired and checked by reading, not compiled, not run on hardware
//
#include <hip/hip_runtime.h>
#include <math.h>

typedef __bf16 bf16_t;
typedef __attribute__((ext_vector_type(16))) __bf16 v16bf;
typedef __attribute__((ext_vector_type(8)))  __bf16 v8bf;
typedef __attribute__((ext_vector_type(8)))  float  v8f;

#define NNODES 150000
#define DD 256

// ---------- WMMA fragment helpers (contiguous k order in A and B, wave32) ----------

// A-matrix 16x32 bf16 fragment from a row-major tile in LDS.
// lane 0-15 : M=lane,    K = k0+0..15
// lane 16-31: M=lane-16, K = k0+16..31
__device__ inline v16bf load_a_frag(const bf16_t* tile, int ldk, int k0, int lane) {
  const int m   = lane & 15;
  const int klo = k0 + ((lane & 16) ? 16 : 0);
  const bf16_t* row = tile + m * ldk + klo;
  v8bf lo = *(const v8bf*)(row);
  v8bf hi = *(const v8bf*)(row + 8);
  return __builtin_shufflevector(lo, hi, 0,1,2,3,4,5,6,7,8,9,10,11,12,13,14,15);
}

// B-matrix 32x16 bf16 fragment. B[k][n] = Wrow(nb+n)[k] (we compute x @ W^T),
// so each lane reads 16 consecutive bf16 from one weight row: contiguous 32B.
__device__ inline v16bf load_b_frag(const bf16_t* w, int ldw, int nb, int k0, int lane) {
  const int n  = lane & 15;
  const int kb = k0 + ((lane & 16) ? 16 : 0);
  return *(const v16bf*)(w + (size_t)(nb + n) * ldw + kb);
}

__device__ inline v8f wmma_bf16(v16bf a, v16bf b, v8f c) {
  return __builtin_amdgcn_wmma_f32_16x16x32_bf16(false, a, false, b, (short)0, c, false, false);
}

__device__ inline void cvt4(bf16_t* d, float4 v) {
  d[0] = (bf16_t)v.x; d[1] = (bf16_t)v.y; d[2] = (bf16_t)v.z; d[3] = (bf16_t)v.w;
}

// ---------- weight preparation ----------

__global__ __launch_bounds__(256) void k_cvt(const float* __restrict__ s,
                                             bf16_t* __restrict__ d, int n) {
  int i = blockIdx.x * 256 + threadIdx.x;
  if (i < n) d[i] = (bf16_t)s[i];
}

// Wrz[j][k] : j in [0,512) = r/z gate rows; k<256 -> Wih[j][k], k>=256 -> Whh[j][k-256]
__global__ __launch_bounds__(256) void k_build_rz(const float* __restrict__ Wih,
                                                  const float* __restrict__ Whh,
                                                  bf16_t* __restrict__ wrz) {
  int i = blockIdx.x * 256 + threadIdx.x;
  if (i < 512 * 512) {
    int j = i >> 9, k = i & 511;
    float v = (k < 256) ? Wih[j * 256 + k] : Whh[j * 256 + (k - 256)];
    wrz[i] = (bf16_t)v;
  }
}

// ---------- p = h @ W^T + b   ([N,256] x [256,256]), M=32 per block ----------

__global__ __launch_bounds__(256) void k_msg_gemm(const float* __restrict__ h,
                                                  const bf16_t* __restrict__ wbf,
                                                  const float* __restrict__ bias,
                                                  float* __restrict__ p) {
  __shared__ __align__(16) bf16_t hbf[32][DD];
  const int tid = threadIdx.x;
  const int m0  = blockIdx.x * 32;
  for (int i = tid; i < 32 * 64; i += 256) {
    int m = i >> 6, c = i & 63;
    int row = m0 + m;
    float4 v = make_float4(0.f, 0.f, 0.f, 0.f);
    if (row < NNODES) v = ((const float4*)(h + (size_t)row * DD))[c];
    cvt4(&hbf[m][c * 4], v);
  }
  __syncthreads();
  const int lane = tid & 31, wave = tid >> 5;
  for (int it = 0; it < 2; ++it) {
    const int nb = (wave * 2 + it) * 16;
    v8f acc0 = {}, acc1 = {};
#pragma unroll
    for (int kc = 0; kc < DD / 32; ++kc) {
      v16bf A0 = load_a_frag(&hbf[0][0],  DD, kc * 32, lane);
      v16bf A1 = load_a_frag(&hbf[16][0], DD, kc * 32, lane);
      v16bf B  = load_b_frag(wbf, DD, nb, kc * 32, lane);
      acc0 = wmma_bf16(A0, B, acc0);
      acc1 = wmma_bf16(A1, B, acc1);
    }
    const int n  = nb + (lane & 15);
    const int mb = (lane & 16) ? 8 : 0;
    const float bv = bias[n];
#pragma unroll
    for (int r = 0; r < 8; ++r) {
      int row0 = m0 + mb + r;
      int row1 = row0 + 16;
      if (row0 < NNODES) p[(size_t)row0 * DD + n] = acc0[r] + bv;
      if (row1 < NNODES) p[(size_t)row1 * DD + n] = acc1[r] + bv;
    }
  }
}

// ---------- a[dst] += p[src]  (edge scatter, f32 HW atomics) ----------

__global__ __launch_bounds__(256) void k_scatter(const float* __restrict__ p,
                                                 const int* __restrict__ src,
                                                 const int* __restrict__ dst,
                                                 float* __restrict__ a, int E) {
  int e = blockIdx.x * 4 + (threadIdx.x >> 6);
  if (e >= E) return;
  int t = threadIdx.x & 63;
  int s = src[e], d = dst[e];
  float4 v = ((const float4*)(p + (size_t)s * DD))[t];
  float* ap = a + (size_t)d * DD + (size_t)t * 4;
  unsafeAtomicAdd(ap + 0, v.x);
  unsafeAtomicAdd(ap + 1, v.y);
  unsafeAtomicAdd(ap + 2, v.z);
  unsafeAtomicAdd(ap + 3, v.w);
}

// ---------- fused GRU, M=32, gating fully in registers ----------
// One wave owns all 4 gate tiles (r, z, i_n, h_n) for its 16-column slice:
// r/z use Wrz over [a|h] (K=512); i_n uses Win over a (kc<8); h_n uses Whn
// over h (kc>=8). The same A fragments feed all gates, and the C-layout
// guarantees r/z/i_n/h_n of one output element share lane+slot -> no LDS
// exchange, no second barrier.

__global__ __launch_bounds__(256) void k_gru(const float* __restrict__ a,
                                             const float* __restrict__ h_in,
                                             float* __restrict__ h_out,
                                             const bf16_t* __restrict__ wrz,
                                             const bf16_t* __restrict__ win,
                                             const bf16_t* __restrict__ whn,
                                             const float* __restrict__ bih,
                                             const float* __restrict__ bhh) {
  __shared__ __align__(16) bf16_t ah[32][512];  // cols 0..255 = a, 256..511 = h
  const int tid = threadIdx.x;
  const int m0  = blockIdx.x * 32;

  for (int i = tid; i < 32 * 64; i += 256) {
    int m = i >> 6, c = i & 63;
    int row = m0 + m;
    float4 va = make_float4(0.f, 0.f, 0.f, 0.f), vh = va;
    if (row < NNODES) {
      va = ((const float4*)(a    + (size_t)row * DD))[c];
      vh = ((const float4*)(h_in + (size_t)row * DD))[c];
    }
    cvt4(&ah[m][c * 4], va);
    cvt4(&ah[m][256 + c * 4], vh);
  }
  __syncthreads();

  const int lane = tid & 31, wave = tid >> 5;
  for (int it = 0; it < 2; ++it) {
    const int nb = (wave * 2 + it) * 16;
    v8f aR[2] = {{}, {}}, aZ[2] = {{}, {}}, aI[2] = {{}, {}}, aH[2] = {{}, {}};
    // kc 0..7: A chunks cover the a-half -> accumulate r, z, i_n
#pragma unroll 2
    for (int kc = 0; kc < 8; ++kc) {
      v16bf A0 = load_a_frag(&ah[0][0],  512, kc * 32, lane);
      v16bf A1 = load_a_frag(&ah[16][0], 512, kc * 32, lane);
      v16bf Br = load_b_frag(wrz, 512, nb,       kc * 32, lane);
      aR[0] = wmma_bf16(A0, Br, aR[0]);
      aR[1] = wmma_bf16(A1, Br, aR[1]);
      v16bf Bz = load_b_frag(wrz, 512, 256 + nb, kc * 32, lane);
      aZ[0] = wmma_bf16(A0, Bz, aZ[0]);
      aZ[1] = wmma_bf16(A1, Bz, aZ[1]);
      v16bf Bi = load_b_frag(win, 256, nb,       kc * 32, lane);
      aI[0] = wmma_bf16(A0, Bi, aI[0]);
      aI[1] = wmma_bf16(A1, Bi, aI[1]);
    }
    // kc 8..15: A chunks cover the h-half -> accumulate r, z, h_n
#pragma unroll 2
    for (int kc = 8; kc < 16; ++kc) {
      v16bf A0 = load_a_frag(&ah[0][0],  512, kc * 32, lane);
      v16bf A1 = load_a_frag(&ah[16][0], 512, kc * 32, lane);
      v16bf Br = load_b_frag(wrz, 512, nb,       kc * 32, lane);
      aR[0] = wmma_bf16(A0, Br, aR[0]);
      aR[1] = wmma_bf16(A1, Br, aR[1]);
      v16bf Bz = load_b_frag(wrz, 512, 256 + nb, kc * 32, lane);
      aZ[0] = wmma_bf16(A0, Bz, aZ[0]);
      aZ[1] = wmma_bf16(A1, Bz, aZ[1]);
      v16bf Bh = load_b_frag(whn, 256, nb, (kc - 8) * 32, lane);
      aH[0] = wmma_bf16(A0, Bh, aH[0]);
      aH[1] = wmma_bf16(A1, Bh, aH[1]);
    }
    // register-resident gating
    const int n  = nb + (lane & 15);
    const int mb = (lane & 16) ? 8 : 0;
    const float br = bih[n]       + bhh[n];
    const float bz = bih[256 + n] + bhh[256 + n];
    const float bi = bih[512 + n];
    const float bh = bhh[512 + n];
#pragma unroll
    for (int half = 0; half < 2; ++half) {
#pragma unroll
      for (int r = 0; r < 8; ++r) {
        int row = m0 + half * 16 + mb + r;
        if (row < NNODES) {
          float rr = 1.f / (1.f + expf(-(aR[half][r] + br)));
          float zz = 1.f / (1.f + expf(-(aZ[half][r] + bz)));
          float nn = tanhf(aI[half][r] + bi + rr * (aH[half][r] + bh));
          float ho = h_in[(size_t)row * DD + n];
          h_out[(size_t)row * DD + n] = (1.f - zz) * nn + zz * ho;
        }
      }
    }
  }
}

// ---------- out = concat([X, Y]) @ w^T + b   ([N,512] x [512,256]), M=32 ----------

__global__ __launch_bounds__(256) void k_concat_fc(const float* __restrict__ X,
                                                   const float* __restrict__ Y,
                                                   const bf16_t* __restrict__ wbf,
                                                   const float* __restrict__ bias,
                                                   float* __restrict__ out) {
  __shared__ __align__(16) bf16_t xy[32][512];
  const int tid = threadIdx.x;
  const int m0  = blockIdx.x * 32;
  for (int i = tid; i < 32 * 64; i += 256) {
    int m = i >> 6, c = i & 63;
    int row = m0 + m;
    float4 vx = make_float4(0.f, 0.f, 0.f, 0.f), vy = vx;
    if (row < NNODES) {
      vx = ((const float4*)(X + (size_t)row * DD))[c];
      vy = ((const float4*)(Y + (size_t)row * DD))[c];
    }
    cvt4(&xy[m][c * 4], vx);
    cvt4(&xy[m][256 + c * 4], vy);
  }
  __syncthreads();
  const int lane = tid & 31, wave = tid >> 5;
  for (int it = 0; it < 2; ++it) {
    const int nb = (wave * 2 + it) * 16;
    v8f acc0 = {}, acc1 = {};
#pragma unroll 4
    for (int kc = 0; kc < 512 / 32; ++kc) {
      v16bf A0 = load_a_frag(&xy[0][0],  512, kc * 32, lane);
      v16bf A1 = load_a_frag(&xy[16][0], 512, kc * 32, lane);
      v16bf B  = load_b_frag(wbf, 512, nb, kc * 32, lane);
      acc0 = wmma_bf16(A0, B, acc0);
      acc1 = wmma_bf16(A1, B, acc1);
    }
    const int n  = nb + (lane & 15);
    const int mb = (lane & 16) ? 8 : 0;
    const float bv = bias[n];
#pragma unroll
    for (int r = 0; r < 8; ++r) {
      int row0 = m0 + mb + r;
      int row1 = row0 + 16;
      if (row0 < NNODES) out[(size_t)row0 * DD + n] = acc0[r] + bv;
      if (row1 < NNODES) out[(size_t)row1 * DD + n] = acc1[r] + bv;
    }
  }
}

// ---------- launcher ----------

extern "C" void kernel_launch(void* const* d_in, const int* in_sizes, int n_in,
                              void* d_out, int out_size, void* d_ws, size_t ws_size,
                              hipStream_t stream) {
  (void)in_sizes; (void)n_in; (void)out_size; (void)ws_size;
  const float* nodes   = (const float*)d_in[0];
  const int*   ast_src = (const int*)d_in[1];
  const int*   ast_dst = (const int*)d_in[2];
  const int*   cfg_src = (const int*)d_in[3];
  const int*   cfg_dst = (const int*)d_in[4];
  const float* W_ast   = (const float*)d_in[5];
  const float* b_ast   = (const float*)d_in[6];
  const float* Wih_ast = (const float*)d_in[7];
  const float* Whh_ast = (const float*)d_in[8];
  const float* bih_ast = (const float*)d_in[9];
  const float* bhh_ast = (const float*)d_in[10];
  const float* w1      = (const float*)d_in[11];
  const float* b1      = (const float*)d_in[12];
  const float* W_cpg   = (const float*)d_in[13];
  const float* b_cpg   = (const float*)d_in[14];
  const float* Wih_cpg = (const float*)d_in[15];
  const float* Whh_cpg = (const float*)d_in[16];
  const float* bih_cpg = (const float*)d_in[17];
  const float* bhh_cpg = (const float*)d_in[18];
  const float* w2      = (const float*)d_in[19];
  const float* b2      = (const float*)d_in[20];

  const size_t NP = (size_t)NNODES * DD;
  float* p    = (float*)d_ws;
  float* aagg = p + NP;
  float* hbuf = aagg + NP;
  float* hidd = hbuf + NP;
  bf16_t* wreg    = (bf16_t*)(hidd + NP);
  bf16_t* Wast_bf = wreg;
  bf16_t* Wrz_ast = Wast_bf + 65536;
  bf16_t* Win_ast = Wrz_ast + 262144;
  bf16_t* Whn_ast = Win_ast + 65536;
  bf16_t* Wcpg_bf = Whn_ast + 65536;
  bf16_t* Wrz_cpg = Wcpg_bf + 65536;
  bf16_t* Win_cpg = Wrz_cpg + 262144;
  bf16_t* Whn_cpg = Win_cpg + 65536;
  bf16_t* w1bf    = Whn_cpg + 65536;
  bf16_t* w2bf    = w1bf + 131072;

  // weight prep (cheap, ~2.4MB)
  k_cvt<<<(65536 + 255) / 256, 256, 0, stream>>>(W_ast, Wast_bf, 65536);
  k_build_rz<<<(512 * 512 + 255) / 256, 256, 0, stream>>>(Wih_ast, Whh_ast, Wrz_ast);
  k_cvt<<<(65536 + 255) / 256, 256, 0, stream>>>(Wih_ast + 512 * 256, Win_ast, 65536);
  k_cvt<<<(65536 + 255) / 256, 256, 0, stream>>>(Whh_ast + 512 * 256, Whn_ast, 65536);
  k_cvt<<<(65536 + 255) / 256, 256, 0, stream>>>(W_cpg, Wcpg_bf, 65536);
  k_build_rz<<<(512 * 512 + 255) / 256, 256, 0, stream>>>(Wih_cpg, Whh_cpg, Wrz_cpg);
  k_cvt<<<(65536 + 255) / 256, 256, 0, stream>>>(Wih_cpg + 512 * 256, Win_cpg, 65536);
  k_cvt<<<(65536 + 255) / 256, 256, 0, stream>>>(Whh_cpg + 512 * 256, Whn_cpg, 65536);
  k_cvt<<<(131072 + 255) / 256, 256, 0, stream>>>(w1, w1bf, 131072);
  k_cvt<<<(131072 + 255) / 256, 256, 0, stream>>>(w2, w2bf, 131072);

  const int GB32 = (NNODES + 31) / 32;  // 4688

  // AST branch (h0 = nodes)
  {
    const float* h = nodes;
    for (int s = 0; s < 2; ++s) {
      k_msg_gemm<<<GB32, 256, 0, stream>>>(h, Wast_bf, b_ast, p);
      hipMemsetAsync(aagg, 0, NP * sizeof(float), stream);
      k_scatter<<<(300000 + 3) / 4, 256, 0, stream>>>(p, ast_src, ast_dst, aagg, 300000);
      k_gru<<<GB32, 256, 0, stream>>>(aagg, h, hbuf, Wrz_ast, Win_ast, Whn_ast, bih_ast, bhh_ast);
      h = hbuf;
    }
  }
  // hiddens = [h_ast | nodes] @ w1^T + b1
  k_concat_fc<<<GB32, 256, 0, stream>>>(hbuf, nodes, w1bf, b1, hidd);
  // CPG branch (h0 = hiddens)
  {
    const float* h = hidd;
    for (int s = 0; s < 2; ++s) {
      k_msg_gemm<<<GB32, 256, 0, stream>>>(h, Wcpg_bf, b_cpg, p);
      hipMemsetAsync(aagg, 0, NP * sizeof(float), stream);
      k_scatter<<<(250000 + 3) / 4, 256, 0, stream>>>(p, cfg_src, cfg_dst, aagg, 250000);
      k_gru<<<GB32, 256, 0, stream>>>(aagg, h, hbuf, Wrz_cpg, Win_cpg, Whn_cpg, bih_cpg, bhh_cpg);
      h = hbuf;
    }
  }
  // logits = [h_cpg | hiddens] @ w2^T + b2
  k_concat_fc<<<GB32, 256, 0, stream>>>(hbuf, hidd, w2bf, b2, (float*)d_out);
}
